// Nonlocal_77232101916722
// MI455X (gfx1250) — matrix operand order repaired and checked by reading, not compiled, not run on hardware
//
#include <hip/hip_runtime.h>
#include <hip/hip_bf16.h>

typedef __attribute__((ext_vector_type(16))) _Float16 v16h;
typedef __attribute__((ext_vector_type(8)))  float    v8f;
typedef __attribute__((ext_vector_type(2)))  _Float16 h2;

constexpr int kB  = 4;
constexpr int kC  = 256;
constexpr int kCI = 128;
constexpr int kHW = 4096;
#define LOG2E 1.44269504088896340736f

__device__ __forceinline__ v8f zero8() {
  v8f z = {0.f, 0.f, 0.f, 0.f, 0.f, 0.f, 0.f, 0.f};
  return z;
}

// A-operand (16x32 f16, M=lane%16): elements 0..15 = row[half*16..+15]
// (contiguous K order, matching load_b16)
__device__ __forceinline__ v16h load_a16(const _Float16* row, int hf) {
  v16h a;
  reinterpret_cast<uint4*>(&a)[0] = *reinterpret_cast<const uint4*>(row + hf * 16);
  reinterpret_cast<uint4*>(&a)[1] = *reinterpret_cast<const uint4*>(row + hf * 16 + 8);
  return a;
}
// B-operand (32x16 f16, N=lane%16): 16 contiguous K-values starting at p
// (caller already applied the half*16 K offset)
__device__ __forceinline__ v16h load_b16(const _Float16* p) {
  v16h b;
  reinterpret_cast<uint4*>(&b)[0] = *reinterpret_cast<const uint4*>(p);
  reinterpret_cast<uint4*>(&b)[1] = *reinterpret_cast<const uint4*>(p + 8);
  return b;
}
// A-operand gathered from an fp32 row (weights), converting to f16
__device__ __forceinline__ v16h load_a_f32(const float* row, int hf) {
  v16h a;
#pragma unroll
  for (int i = 0; i < 8; ++i) a[i] = (_Float16)row[hf * 16 + i];
#pragma unroll
  for (int i = 0; i < 8; ++i) a[8 + i] = (_Float16)row[hf * 16 + 8 + i];
  return a;
}
__device__ __forceinline__ v8f wmma16(v16h a, v16h b, v8f c) {
  return __builtin_amdgcn_wmma_f32_16x16x32_f16(false, a, false, b, (short)0, c,
                                                false, false);
}

// ---------------------------------------------------------------------------
// Kernel 1: fused theta/phi/g projections.
//   Q = theta proj, stored (B, HW, CI) f16   (row = query, contiguous channels)
//   K = phi   proj, stored (B, HW, CI) f16   (row = key,   contiguous channels)
//   V = g     proj, stored (B, CI, HW) f16   (row = channel, contiguous keys)
// Block: 256 thr (8 waves), handles one (batch, 64-wide hw tile).
// Wave w computes ci rows [16w, 16w+16) for all three projections, sharing B.
// ---------------------------------------------------------------------------
__global__ __launch_bounds__(256) void proj_kernel(
    const float* __restrict__ x,
    const float* __restrict__ thw, const float* __restrict__ thb,
    const float* __restrict__ phw, const float* __restrict__ phb,
    const float* __restrict__ gw,  const float* __restrict__ gb,
    _Float16* __restrict__ Qb, _Float16* __restrict__ Kb,
    _Float16* __restrict__ Vt) {
  __shared__ _Float16 Xl[64][kC + 8];   // x tile transposed: [hw_local][channel]
  const int tid   = threadIdx.x;
  const int batch = blockIdx.x >> 6;
  const int n0    = (blockIdx.x & 63) * 64;

  {  // stage X (coalesced along hw), write transposed as f16
    const int nl = tid & 63, c0 = tid >> 6;
    const float* xb = x + (size_t)batch * kC * kHW + n0 + nl;
    for (int c = c0; c < kC; c += 4)
      Xl[nl][c] = (_Float16)xb[(size_t)c * kHW];
  }
  __syncthreads();

  const int wid = tid >> 5, lane = tid & 31, l16 = lane & 15, hf = lane >> 4;
  const int m0 = wid * 16;

  v8f acc0[4], acc1[4], acc2[4];
#pragma unroll
  for (int j = 0; j < 4; ++j) { acc0[j] = zero8(); acc1[j] = zero8(); acc2[j] = zero8(); }

  const float* r0 = thw + (size_t)(m0 + l16) * kC;
  const float* r1 = phw + (size_t)(m0 + l16) * kC;
  const float* r2 = gw  + (size_t)(m0 + l16) * kC;

  for (int cc = 0; cc < 8; ++cc) {  // K = 256 channels, 32 per step
    v16h a0 = load_a_f32(r0 + cc * 32, hf);
    v16h a1 = load_a_f32(r1 + cc * 32, hf);
    v16h a2 = load_a_f32(r2 + cc * 32, hf);
#pragma unroll
    for (int j = 0; j < 4; ++j) {
      v16h bb = load_b16(&Xl[j * 16 + l16][cc * 32 + hf * 16]);
      acc0[j] = wmma16(a0, bb, acc0[j]);
      acc1[j] = wmma16(a1, bb, acc1[j]);
      acc2[j] = wmma16(a2, bb, acc2[j]);
    }
  }

  // Bias + stores. C-layout: reg r <-> row m0 + hf*8 + r, col n0 + j*16 + l16.
#pragma unroll
  for (int j = 0; j < 4; ++j) {
    const int n = n0 + j * 16 + l16;
#pragma unroll
    for (int t = 0; t < 4; ++t) {  // adjacent row pairs -> packed dword stores
      const int m = m0 + hf * 8 + 2 * t;
      h2 q2, k2;
      q2[0] = (_Float16)(acc0[j][2 * t]     + thb[m]);
      q2[1] = (_Float16)(acc0[j][2 * t + 1] + thb[m + 1]);
      k2[0] = (_Float16)(acc1[j][2 * t]     + phb[m]);
      k2[1] = (_Float16)(acc1[j][2 * t + 1] + phb[m + 1]);
      *reinterpret_cast<h2*>(Qb + ((size_t)batch * kHW + n) * kCI + m) = q2;
      *reinterpret_cast<h2*>(Kb + ((size_t)batch * kHW + n) * kCI + m) = k2;
    }
#pragma unroll
    for (int r = 0; r < 8; ++r) {
      const int m = m0 + hf * 8 + r;
      Vt[((size_t)batch * kCI + m) * kHW + n] = (_Float16)(acc2[j][r] + gb[m]);
    }
  }
}

// ---------------------------------------------------------------------------
// Kernel 2: flash attention. One wave per 16-query tile; loop over 32-key
// tiles with online softmax. QK: A=Q(16x32ch), B=K(32ch x 16key) x2 key groups.
// PV: A=P(16x32key) via LDS transpose, B=V(32key x 16ci) x8 ci chunks.
// ---------------------------------------------------------------------------
__global__ __launch_bounds__(128) void attn_kernel(
    const _Float16* __restrict__ Qb, const _Float16* __restrict__ Kbp,
    const _Float16* __restrict__ Vt, _Float16* __restrict__ Ob) {
  __shared__ _Float16 Pl[4][16][48];  // per-wave P tile, 48-stride = 16B aligned
  const int tid = threadIdx.x, wid = tid >> 5, lane = tid & 31;
  const int l16 = lane & 15, hf = lane >> 4;
  const int wg = blockIdx.x * 4 + wid;   // 1024 waves total
  const int batch = wg >> 8;
  const int q0 = (wg & 255) * 16;

  v16h Qa[4];
  const _Float16* qrow = Qb + ((size_t)batch * kHW + q0 + l16) * kCI;
#pragma unroll
  for (int cc = 0; cc < 4; ++cc) Qa[cc] = load_a16(qrow + cc * 32, hf);

  float mrow[8], lrow[8];
  v8f O[8];
#pragma unroll
  for (int r = 0; r < 8; ++r) { mrow[r] = -3.0e38f; lrow[r] = 0.f; }
#pragma unroll
  for (int j = 0; j < 8; ++j) O[j] = zero8();

  const _Float16* kbase = Kbp + (size_t)batch * kHW * kCI;
  const _Float16* vbase = Vt  + (size_t)batch * kCI * kHW;

  for (int kt = 0; kt < 128; ++kt) {
    const int k0 = kt * 32;
    const _Float16* kr0 = kbase + (size_t)(k0 + l16) * kCI + hf * 16;
    const _Float16* kr1 = kbase + (size_t)(k0 + 16 + l16) * kCI + hf * 16;
    v8f s0 = zero8(), s1 = zero8();
#pragma unroll
    for (int cc = 0; cc < 4; ++cc) {
      v16h b0 = load_b16(kr0 + cc * 32);
      v16h b1 = load_b16(kr1 + cc * 32);
      s0 = wmma16(Qa[cc], b0, s0);
      s1 = wmma16(Qa[cc], b1, s1);
    }
    // online softmax over the 32 key columns (xor 1/2/4/8 stays in 16-lane half)
    float p0[8], p1[8];
#pragma unroll
    for (int r = 0; r < 8; ++r) {
      float cm = fmaxf(s0[r], s1[r]);
      cm = fmaxf(cm, __shfl_xor(cm, 1, 32));
      cm = fmaxf(cm, __shfl_xor(cm, 2, 32));
      cm = fmaxf(cm, __shfl_xor(cm, 4, 32));
      cm = fmaxf(cm, __shfl_xor(cm, 8, 32));
      const float mn = fmaxf(mrow[r], cm);
      const float sc = exp2f((mrow[r] - mn) * LOG2E);
      p0[r] = exp2f((s0[r] - mn) * LOG2E);
      p1[r] = exp2f((s1[r] - mn) * LOG2E);
      float rs = p0[r] + p1[r];
      rs += __shfl_xor(rs, 1, 32);
      rs += __shfl_xor(rs, 2, 32);
      rs += __shfl_xor(rs, 4, 32);
      rs += __shfl_xor(rs, 8, 32);
      lrow[r] = lrow[r] * sc + rs;
      mrow[r] = mn;
#pragma unroll
      for (int j = 0; j < 8; ++j) O[j][r] *= sc;
    }
    // C-layout -> A-layout transpose of P through per-wave LDS
#pragma unroll
    for (int r = 0; r < 8; ++r) {
      Pl[wid][hf * 8 + r][l16]      = (_Float16)p0[r];
      Pl[wid][hf * 8 + r][16 + l16] = (_Float16)p1[r];
    }
    asm volatile("s_wait_dscnt 0" ::: "memory");  // LDS RAW, same wave, in-order
    v16h Pa = load_a16(&Pl[wid][l16][0], hf);
#pragma unroll
    for (int j = 0; j < 8; ++j) {  // V tile: row = ci channel, contiguous keys
      v16h bv = load_b16(vbase + (size_t)(j * 16 + l16) * kHW + k0 + hf * 16);
      O[j] = wmma16(Pa, bv, O[j]);
    }
    asm volatile("" ::: "memory");  // keep next-iter P stores after Pa loads
  }
  // normalize and store O as (B, HW, CI) f16 for the out-projection GEMM
#pragma unroll
  for (int r = 0; r < 8; ++r) {
    const float inv = 1.0f / lrow[r];
#pragma unroll
    for (int j = 0; j < 8; ++j) O[j][r] *= inv;
  }
#pragma unroll
  for (int r = 0; r < 8; ++r) {
    _Float16* orow = Ob + ((size_t)batch * kHW + q0 + hf * 8 + r) * kCI + l16;
#pragma unroll
    for (int j = 0; j < 8; ++j) orow[j * 16] = (_Float16)O[j][r];
  }
}

// ---------------------------------------------------------------------------
// Kernel 3: out2[b,c,q] = out_w(256x128) @ O^T + bias, fp32 (B, C, HW).
// Block: 8 waves, one (batch, 64-wide q tile); each wave does 2 c-row tiles.
// ---------------------------------------------------------------------------
__global__ __launch_bounds__(256) void outproj_kernel(
    const _Float16* __restrict__ Ob, const float* __restrict__ ow,
    const float* __restrict__ obias, float* __restrict__ out2) {
  const int tid = threadIdx.x;
  const int batch = blockIdx.x >> 6;
  const int n0 = (blockIdx.x & 63) * 64;
  const int wid = tid >> 5, lane = tid & 31, l16 = lane & 15, hf = lane >> 4;
  const int m0a = wid * 16, m0b = m0a + 128;

  v8f acc0[4], acc1[4];
#pragma unroll
  for (int j = 0; j < 4; ++j) { acc0[j] = zero8(); acc1[j] = zero8(); }
  const float* ra = ow + (size_t)(m0a + l16) * kCI;
  const float* rb = ow + (size_t)(m0b + l16) * kCI;
#pragma unroll
  for (int cc = 0; cc < 4; ++cc) {  // K = CI = 128
    v16h a0 = load_a_f32(ra + cc * 32, hf);
    v16h a1 = load_a_f32(rb + cc * 32, hf);
#pragma unroll
    for (int j = 0; j < 4; ++j) {
      v16h bb = load_b16(Ob + ((size_t)batch * kHW + n0 + j * 16 + l16) * kCI +
                         cc * 32 + hf * 16);
      acc0[j] = wmma16(a0, bb, acc0[j]);
      acc1[j] = wmma16(a1, bb, acc1[j]);
    }
  }
#pragma unroll
  for (int t = 0; t < 2; ++t) {
    const int m0 = t ? m0b : m0a;
    v8f* acc = t ? acc1 : acc0;
#pragma unroll
    for (int r = 0; r < 8; ++r) {
      const int c = m0 + hf * 8 + r;
      const float bi = obias[c];
      float* orow = out2 + ((size_t)batch * kC + c) * kHW + n0 + l16;
#pragma unroll
      for (int j = 0; j < 4; ++j) orow[j * 16] = acc[j][r] + bi;
    }
  }
}

// ---------------------------------------------------------------------------
// Kernel 4: deterministic BN stats per channel; fold into y = v*a[c] + b[c].
// ---------------------------------------------------------------------------
__global__ __launch_bounds__(256) void bn_stats_kernel(
    const float* __restrict__ out2, const float* __restrict__ gamma,
    const float* __restrict__ beta, float* __restrict__ ab) {
  __shared__ float ss[256], sq[256];
  const int c = blockIdx.x, tid = threadIdx.x;
  float s = 0.f, q = 0.f;
#pragma unroll
  for (int b = 0; b < kB; ++b) {
    const float* p = out2 + ((size_t)b * kC + c) * kHW;
    for (int i = tid; i < kHW; i += 256) { float v = p[i]; s += v; q += v * v; }
  }
  ss[tid] = s; sq[tid] = q;
  __syncthreads();
  for (int off = 128; off > 0; off >>= 1) {
    if (tid < off) { ss[tid] += ss[tid + off]; sq[tid] += sq[tid + off]; }
    __syncthreads();
  }
  if (tid == 0) {
    const float invN = 1.0f / (float)(kB * kHW);
    const float mean = ss[0] * invN;
    const float var  = sq[0] * invN - mean * mean;
    const float a = gamma[c] * rsqrtf(var + 1e-5f);
    ab[c] = a;
    ab[kC + c] = beta[c] - mean * a;
  }
}

// ---------------------------------------------------------------------------
// Kernel 5: y = relu(out2*a + b) + x
// ---------------------------------------------------------------------------
__global__ __launch_bounds__(256) void epilogue_kernel(
    const float* __restrict__ out2, const float* __restrict__ x,
    const float* __restrict__ ab, float* __restrict__ y) {
  const size_t i = (size_t)blockIdx.x * 256 + threadIdx.x;
  const int c = (int)((i >> 12) & (kC - 1));  // HW = 4096 = 2^12
  const float v = out2[i] * ab[c] + ab[kC + c];
  y[i] = fmaxf(v, 0.f) + x[i];
}

extern "C" void kernel_launch(void* const* d_in, const int* in_sizes, int n_in,
                              void* d_out, int out_size, void* d_ws,
                              size_t ws_size, hipStream_t stream) {
  const float* x     = (const float*)d_in[0];
  const float* g_w   = (const float*)d_in[1];
  const float* g_b   = (const float*)d_in[2];
  const float* th_w  = (const float*)d_in[3];
  const float* th_b  = (const float*)d_in[4];
  const float* ph_w  = (const float*)d_in[5];
  const float* ph_b  = (const float*)d_in[6];
  const float* out_w = (const float*)d_in[7];
  const float* out_b = (const float*)d_in[8];
  const float* gamma = (const float*)d_in[9];
  const float* beta  = (const float*)d_in[10];

  char* ws = (char*)d_ws;
  const size_t szP = (size_t)kB * kHW * kCI * sizeof(_Float16);  // 4 MB each
  _Float16* Qb = (_Float16*)(ws);
  _Float16* Kb = (_Float16*)(ws + szP);
  _Float16* Vt = (_Float16*)(ws + 2 * szP);
  _Float16* Ob = (_Float16*)(ws + 3 * szP);
  float* out2  = (float*)(ws + 4 * szP);
  float* ab    = (float*)(ws + 4 * szP + (size_t)kB * kC * kHW * sizeof(float));

  proj_kernel<<<kB * 64, 256, 0, stream>>>(x, th_w, th_b, ph_w, ph_b, g_w, g_b,
                                           Qb, Kb, Vt);
  attn_kernel<<<kB * 64, 128, 0, stream>>>(Qb, Kb, Vt, Ob);
  outproj_kernel<<<kB * 64, 256, 0, stream>>>(Ob, out_w, out_b, out2);
  bn_stats_kernel<<<kC, 256, 0, stream>>>(out2, gamma, beta, ab);
  epilogue_kernel<<<(kB * kC * kHW) / 256, 256, 0, stream>>>(out2, x, ab,
                                                             (float*)d_out);
}
